// TRMGATLayer_28389733826666
// MI455X (gfx1250) — matrix operand order repaired and checked by reading, not compiled, not run on hardware
//
#include <hip/hip_runtime.h>

typedef __attribute__((ext_vector_type(16))) _Float16 v16h;
typedef __attribute__((ext_vector_type(8)))  _Float16 v8h;
typedef __attribute__((ext_vector_type(8)))  float    v8f;
typedef __attribute__((ext_vector_type(4)))  unsigned u32x4;
typedef __attribute__((ext_vector_type(8)))  int      i32x8;
typedef __attribute__((ext_vector_type(4)))  int      i32x4;

#define N_NODES 8192
#define FDIM    256

// ---- WMMA operand loaders -------------------------------------------------
// A-matrix (16-bit, 16x32) interleaved layout: lane L (h=L/16, M=L%16),
// element e: K = (e<8 ? h*8 + e : 16 + h*8 + (e-8)) + c0
__device__ __forceinline__ v16h load_a16(const _Float16* row, int c0, int h) {
  v8h lo = *(const v8h*)(row + c0 + h * 8);
  v8h hi = *(const v8h*)(row + c0 + 16 + h * 8);
  v16h r;
#pragma unroll
  for (int i = 0; i < 8; ++i) { r[i] = lo[i]; r[i + 8] = hi[i]; }
  return r;
}

// B-matrix (16-bit, 32x16) interleaved layout: lane L holds column N=L%16;
// element e: K = (e<8 ? h*8 + e : 16 + h*8 + (e-8)) + c0
__device__ __forceinline__ v16h load_b16(const _Float16* row, int c0, int h) {
  v8h lo = *(const v8h*)(row + c0 + h * 8);
  v8h hi = *(const v8h*)(row + c0 + 16 + h * 8);
  v16h r;
#pragma unroll
  for (int i = 0; i < 8; ++i) { r[i] = lo[i]; r[i + 8] = hi[i]; }
  return r;
}

__device__ __forceinline__ v16h load_a16_f32(const float* row, int c0, int h) {
  const float* p0 = row + c0 + h * 8;
  const float* p1 = row + c0 + 16 + h * 8;
  v16h r;
#pragma unroll
  for (int i = 0; i < 8; ++i) { r[i] = (_Float16)p0[i]; r[i + 8] = (_Float16)p1[i]; }
  return r;
}

__device__ __forceinline__ v16h load_b16_f32(const float* row, int c0, int h) {
  const float* p0 = row + c0 + h * 8;
  const float* p1 = row + c0 + 16 + h * 8;
  v16h r;
#pragma unroll
  for (int i = 0; i < 8; ++i) { r[i] = (_Float16)p0[i]; r[i + 8] = (_Float16)p1[i]; }
  return r;
}

// ---- Tensor Data Mover descriptors (CDNA5 D#, ISA 08_async_tensor §8) -----
__device__ __forceinline__ unsigned lds_off(const void* p) {
  // flat LDS aperture: addr[31:0] is the LDS byte address
  return (unsigned)(unsigned long long)p;
}

__device__ __forceinline__ u32x4 tdm_g0(unsigned lds_addr, const void* gptr) {
  unsigned long long ga = (unsigned long long)gptr;
  u32x4 g0;
  g0[0] = 1u;                                                 // count=1, user D#
  g0[1] = lds_addr;                                           // lds_addr[31:0]
  g0[2] = (unsigned)ga;                                       // global_addr[31:0]
  g0[3] = (unsigned)((ga >> 32) & 0x01FFFFFFu) | 0x80000000u; // addr[56:32] | type=2
  return g0;
}

// 1-D copy descriptor, data_size = 8 bytes
__device__ __forceinline__ i32x8 tdm_g1_1d(unsigned tile_elems8,
                                           unsigned long long tensor_elems8) {
  i32x8 g;
  g[0] = 3 << 16;                                   // data_size=3 (8B)
  g[1] = (int)((tensor_elems8 & 0xffffu) << 16);    // tensor_dim0[15:0]
  g[2] = (int)((tensor_elems8 >> 16) & 0xffffu);    // tensor_dim0[31:16]
  g[3] = (int)(tile_elems8 << 16);                  // tile_dim0
  g[4] = 0; g[5] = 0; g[6] = 0; g[7] = 0;           // 1-D: tile_dim1=0
  return g;
}

// 2-D tile descriptor, data_size = 2 bytes (f16)
__device__ __forceinline__ i32x8 tdm_g1_2d(unsigned tile_d0, unsigned tile_d1,
                                           unsigned tensor_d0, unsigned tensor_d1,
                                           unsigned long long stride0) {
  i32x8 g;
  g[0] = 1 << 16;                                                     // data_size=1 (2B)
  g[1] = (int)((tensor_d0 & 0xffffu) << 16);
  g[2] = (int)(((tensor_d0 >> 16) & 0xffffu) | ((tensor_d1 & 0xffffu) << 16));
  g[3] = (int)(((tensor_d1 >> 16) & 0xffffu) | (tile_d0 << 16));
  g[4] = (int)(tile_d1 & 0xffffu);                                    // tile_dim1, tile_dim2=0
  g[5] = (int)(stride0 & 0xffffffffu);                                // dim0_stride[31:0]
  g[6] = (int)((stride0 >> 32) & 0xffffu);                            // dim0_stride[47:32]
  g[7] = 0;
  return g;
}

__device__ __forceinline__ void tdm_load(u32x4 g0, i32x8 g1) {
  i32x4 z4 = {0, 0, 0, 0};
#if defined(__clang_major__) && (__clang_major__ >= 23)
  i32x8 z8 = {0, 0, 0, 0, 0, 0, 0, 0};
  __builtin_amdgcn_tensor_load_to_lds(g0, g1, z4, z4, z8, 0);
#else
  __builtin_amdgcn_tensor_load_to_lds(g0, g1, z4, z4, 0);
#endif
}

// ---- Kernel 1: out = f16( (x @ W^T + b) * scale ), optionally transposed --
__global__ void __launch_bounds__(256)
proj_f16_kernel(const float* __restrict__ x, const float* __restrict__ W,
                const float* __restrict__ bias, _Float16* __restrict__ out,
                float scale, int transpose_out) {
  const int lane = threadIdx.x & 31;
  const int wave = threadIdx.x >> 5;
  const int h    = lane >> 4;
  const int nloc = lane & 15;
  const int tile = blockIdx.x * 8 + wave;      // 16-row tile, 512 total

  v16h A[8];
  const float* xrow = x + (size_t)(tile * 16 + nloc) * FDIM;
#pragma unroll
  for (int kb = 0; kb < 8; ++kb) A[kb] = load_a16_f32(xrow, kb * 32, h);

  for (int nt = 0; nt < 16; ++nt) {
    v8f acc;
#pragma unroll
    for (int i = 0; i < 8; ++i) acc[i] = 0.f;
    const float* wrow = W + (size_t)(nt * 16 + nloc) * FDIM;
#pragma unroll
    for (int kb = 0; kb < 8; ++kb) {
      v16h B = load_b16_f32(wrow, kb * 32, h);
      acc = __builtin_amdgcn_wmma_f32_16x16x32_f16(false, A[kb], false, B,
                                                   (short)0, acc, false, false);
    }
    const int ncol = nt * 16 + nloc;
    const float bv = bias[ncol];
#pragma unroll
    for (int r = 0; r < 8; ++r) {
      float v = (acc[r] + bv) * scale;        // C layout: row = r + 8h
      const int rowg = tile * 16 + r + 8 * h;
      if (!transpose_out) out[(size_t)rowg * FDIM + ncol]    = (_Float16)v;
      else                out[(size_t)ncol * N_NODES + rowg] = (_Float16)v;
    }
  }
}

// ---- Kernel 2: fused masked flash-attention + ELU, TDM double-buffered ----
// dynamic LDS layout (72 KB, well under the 320 KB WGP budget):
//   [0,16K)   K buf0     [16K,32K) Vt buf0
//   [32K,48K) K buf1     [48K,64K) Vt buf1
//   [64K,72K) per-wave P tiles
__global__ void __launch_bounds__(256)
gat_attn_kernel(const _Float16* __restrict__ Q, const _Float16* __restrict__ K,
                const _Float16* __restrict__ Vt, const int* __restrict__ adj,
                float* __restrict__ out) {
  extern __shared__ char smem[];
  _Float16* Kbuf0 = (_Float16*)(smem);
  _Float16* Vbuf0 = (_Float16*)(smem + 16384);
  _Float16* Kbuf1 = (_Float16*)(smem + 32768);
  _Float16* Vbuf1 = (_Float16*)(smem + 49152);
  _Float16* Plds  = (_Float16*)(smem + 65536);

  const int tid  = threadIdx.x;
  const int lane = tid & 31;
  const int wave = tid >> 5;
  const int h    = lane >> 4;
  const int nloc = lane & 15;
  const int tile = blockIdx.x * 8 + wave;
  const int rowbase = tile * 16;
  const int JBLK = N_NODES / 32;

  // resident Q tile (pre-scaled by 1/sqrt(d) in projection)
  v16h Aq[8];
  const _Float16* qrow = Q + (size_t)(rowbase + nloc) * FDIM;
#pragma unroll
  for (int kb = 0; kb < 8; ++kb) Aq[kb] = load_a16(qrow, kb * 32, h);

  v8f acc[16];
#pragma unroll
  for (int ft = 0; ft < 16; ++ft)
#pragma unroll
    for (int i = 0; i < 8; ++i) acc[ft][i] = 0.f;

  float mrow[8], lrow[8];
#pragma unroll
  for (int r = 0; r < 8; ++r) { mrow[r] = -__builtin_inff(); lrow[r] = 0.f; }

  _Float16* pw = Plds + wave * 512;

  const unsigned ldsK[2] = {lds_off(Kbuf0), lds_off(Kbuf1)};
  const unsigned ldsV[2] = {lds_off(Vbuf0), lds_off(Vbuf1)};
  const unsigned long long kTensorElems8 = (unsigned long long)N_NODES * FDIM / 4; // 8B units

  // prologue: TDM-load block 0 into buffer 0 (one wave drives the DMA)
  if (wave == 0) {
    tdm_load(tdm_g0(ldsK[0], K),  tdm_g1_1d(32 * FDIM / 4, kTensorElems8));
    tdm_load(tdm_g0(ldsV[0], Vt), tdm_g1_2d(32, FDIM, N_NODES, FDIM, N_NODES));
  }

  for (int j = 0; j < JBLK; ++j) {
    const int cur = j & 1, nxt = cur ^ 1;
    __syncthreads();                    // all waves done reading buffer `nxt`
    if (wave == 0) {
      if (j + 1 < JBLK) {               // prefetch block j+1 via TDM
        tdm_load(tdm_g0(ldsK[nxt], K + (size_t)(j + 1) * 32 * FDIM),
                 tdm_g1_1d(32 * FDIM / 4, kTensorElems8));
        tdm_load(tdm_g0(ldsV[nxt], Vt + (size_t)(j + 1) * 32),
                 tdm_g1_2d(32, FDIM, N_NODES, FDIM, N_NODES));
        __builtin_amdgcn_s_wait_tensorcnt((short)2);  // block j done, j+1 in flight
      } else {
        __builtin_amdgcn_s_wait_tensorcnt((short)0);
      }
    }
    __syncthreads();                    // buffer `cur` published to all waves
    const _Float16* Kb = cur ? Kbuf1 : Kbuf0;
    const _Float16* Vb = cur ? Vbuf1 : Vbuf0;

    // S tiles: columns [j*32, j*32+16) and [j*32+16, j*32+32)
    v8f s0, s1;
#pragma unroll
    for (int i = 0; i < 8; ++i) { s0[i] = 0.f; s1[i] = 0.f; }
#pragma unroll
    for (int kb = 0; kb < 8; ++kb) {
      v16h b0 = load_b16(Kb + (size_t)nloc * FDIM,        kb * 32, h);
      v16h b1 = load_b16(Kb + (size_t)(16 + nloc) * FDIM, kb * 32, h);
      s0 = __builtin_amdgcn_wmma_f32_16x16x32_f16(false, Aq[kb], false, b0,
                                                  (short)0, s0, false, false);
      s1 = __builtin_amdgcn_wmma_f32_16x16x32_f16(false, Aq[kb], false, b1,
                                                  (short)0, s1, false, false);
    }

    // adjacency mask (non-temporal: 256 MB single-use stream, keep L2 for Q/K/V)
    // + online softmax (rows striped as r+8h across C VGPRs)
    const int c0 = j * 32 + nloc;
    float alpha[8];
#pragma unroll
    for (int r = 0; r < 8; ++r) {
      const size_t arow = (size_t)(rowbase + r + 8 * h) * N_NODES;
      const int a0 = __builtin_nontemporal_load(adj + arow + c0);
      const int a1 = __builtin_nontemporal_load(adj + arow + c0 + 16);
      const float v0 = (a0 > 0) ? s0[r] : -9e15f;
      const float v1 = (a1 > 0) ? s1[r] : -9e15f;
      float t = fmaxf(v0, v1);
#pragma unroll
      for (int off = 1; off < 16; off <<= 1) t = fmaxf(t, __shfl_xor(t, off, 32));
      const float mnew = fmaxf(mrow[r], t);
      const float a  = __expf(mrow[r] - mnew);
      const float p0 = __expf(v0 - mnew);
      const float p1 = __expf(v1 - mnew);
      float sum = p0 + p1;
#pragma unroll
      for (int off = 1; off < 16; off <<= 1) sum += __shfl_xor(sum, off, 32);
      lrow[r] = lrow[r] * a + sum;
      mrow[r] = mnew;
      alpha[r] = a;
      pw[(r + 8 * h) * 32 + nloc]      = (_Float16)p0;   // row-major P tile
      pw[(r + 8 * h) * 32 + 16 + nloc] = (_Float16)p1;
    }
#pragma unroll
    for (int ft = 0; ft < 16; ++ft)
#pragma unroll
      for (int r = 0; r < 8; ++r) acc[ft][r] *= alpha[r];

    // per-wave LDS write->read ordering for P
    asm volatile("s_wait_dscnt 0x0" ::: "memory");

    v16h Ap = load_a16(pw + (size_t)nloc * 32, 0, h);
#pragma unroll
    for (int ft = 0; ft < 16; ++ft) {
      v16h Bv = load_b16(Vb + (size_t)(ft * 16 + nloc) * 32, 0, h);
      acc[ft] = __builtin_amdgcn_wmma_f32_16x16x32_f16(false, Ap, false, Bv,
                                                       (short)0, acc[ft], false, false);
    }
  }

  // epilogue: normalize, ELU, non-temporal fp32 store (write-once stream)
#pragma unroll
  for (int ft = 0; ft < 16; ++ft) {
    const int col = ft * 16 + nloc;
#pragma unroll
    for (int r = 0; r < 8; ++r) {
      float v = acc[ft][r] / lrow[r];
      v = (v > 0.f) ? v : expm1f(v);
      __builtin_nontemporal_store(v, out + (size_t)(rowbase + r + 8 * h) * FDIM + col);
    }
  }
}

// ---- Host launch ----------------------------------------------------------
extern "C" void kernel_launch(void* const* d_in, const int* in_sizes, int n_in,
                              void* d_out, int out_size, void* d_ws, size_t ws_size,
                              hipStream_t stream) {
  const float* x   = (const float*)d_in[0];
  const int*   adj = (const int*)d_in[1];
  const float* Wq  = (const float*)d_in[2];
  const float* bq  = (const float*)d_in[3];
  const float* Wk  = (const float*)d_in[4];
  const float* bk  = (const float*)d_in[5];
  const float* Wv  = (const float*)d_in[6];
  const float* bv  = (const float*)d_in[7];
  float* out = (float*)d_out;

  _Float16* qf = (_Float16*)d_ws;                       // 4 MB, row-major, pre-scaled
  _Float16* kf = qf + (size_t)N_NODES * FDIM;           // 4 MB, row-major
  _Float16* vt = kf + (size_t)N_NODES * FDIM;           // 4 MB, transposed [FDIM][N]

  dim3 blk(256);
  dim3 grd(N_NODES / (16 * 8));                         // 64 blocks, 8 waves each

  proj_f16_kernel<<<grd, blk, 0, stream>>>(x, Wq, bq, qf, 1.0f / 16.0f, 0);
  proj_f16_kernel<<<grd, blk, 0, stream>>>(x, Wk, bk, kf, 1.0f, 0);
  proj_f16_kernel<<<grd, blk, 0, stream>>>(x, Wv, bv, vt, 1.0f, 1);
  gat_attn_kernel<<<grd, blk, 73728, stream>>>(qf, kf, vt, adj, out);
}
